// FalconH1HybridAttentionDecoderLayer_82386062672274
// MI455X (gfx1250) — matrix operand order repaired and checked by reading, not compiled, not run on hardware
//
#include <hip/hip_runtime.h>
#include <math.h>

// ---------------- types ----------------
typedef __bf16 bf16_t;
typedef __attribute__((ext_vector_type(16))) __bf16 v16bf;
typedef __attribute__((ext_vector_type(8)))  __bf16 v8bf;
typedef __attribute__((ext_vector_type(8)))  float  v8f;

// ---------------- model constants ----------------
#define T_SEQ   2048
#define D_MODEL 2048
#define NHQ     16
#define NHKV    8
#define DH      128
#define QKV_N   4096          // Q_SIZE + 2*KV_SIZE
#define INTER   8192
#define D_SSM   4096
#define GS      128           // NGROUPS * DSTATE
#define SSM_NH  64
#define HEADD   64            // D_SSM / SSM_NH
#define CONV_CH 4352          // D_SSM + 2*GS
#define ZX_N    8512          // 2*D_SSM + 2*GS + SSM_NH
#define EPS_F   1e-5f

__device__ __forceinline__ float siluf(float x) { return x / (1.f + __expf(-x)); }
__device__ __forceinline__ bf16_t to_bf(float x) { return (bf16_t)x; }

// ============================================================
// Tiled bf16 WMMA GEMM:  C(MxN,f32) = alpha * A(MxK) * B(KxN) [+ C]
// TRANSB=true: B stored row-major as (N x K), computes A * B^T.
//
// 256 threads = 8 waves. Block tile 128x64, K-step 32, double-
// buffered LDS. A tile and transposed-B tile are staged with
// GLOBAL_LOAD_ASYNC_TO_LDS_B128 (ASYNCcnt) so the DMA of tile s+1
// overlaps the ds_load+WMMA of tile s; one s_wait_asynccnt 0 +
// barrier per K-step.
// Waves: 4 M sub-tiles x 2 N sub-tiles; each wave owns 32x32
//   -> 4x v_wmma_f32_16x16x32_bf16 per K-step (A/B frag reuse).
// LDS rows padded to 40 elements (80 B = 20 banks) so half-wave
// b128 fragment loads are bank-conflict-free.
// ============================================================
#define BM 128
#define BN 64
#define BK 32
#define LPAD 40   // padded LDS row length (elements)

#define WMMA_BF16(A_, B_, C_) \
    __builtin_amdgcn_wmma_f32_16x16x32_bf16(false, (A_), false, (B_), \
                                            (short)0, (C_), false, false)

#define ASYNC_B128(ldsoff, gaddr) \
    asm volatile("global_load_async_to_lds_b128 %0, %1, off" \
                 :: "v"(ldsoff), "v"(gaddr) : "memory")
#define WAIT_ASYNC() asm volatile("s_wait_asynccnt 0" ::: "memory")

template <bool TRANSB>
__global__ __launch_bounds__(256) void gemm_wmma_bf16(
    const bf16_t* __restrict__ A, int lda,
    const bf16_t* __restrict__ B, int ldb,
    float* __restrict__ C, int ldc,
    int M, int N, int K, float alpha, int accum)
{
    __shared__ __align__(16) bf16_t As[2][BM * LPAD];   // 2 x 10 KB
    __shared__ __align__(16) bf16_t Bs[2][BN * LPAD];   // 2 x  5 KB

    const int m0   = blockIdx.y * BM;
    const int n0   = blockIdx.x * BN;
    const int tid  = threadIdx.x;
    const int wv   = tid >> 5;
    const int lane = tid & 31;
    const int wm   = wv & 3;      // 4 M sub-tiles of 32
    const int wn   = wv >> 2;     // 2 N sub-tiles of 32
    const int r    = lane & 15;
    const int hi   = lane >> 4;

    // staging indices
    const int smm = tid >> 2;          // 0..63
    const int sch = (tid & 3) * 8;     // K chunk start
    const int bkk = tid >> 3;          // 0..31
    const int bnb = (tid & 7) * 8;     // 8 consecutive n

    v8f acc00 = {0.f,0.f,0.f,0.f,0.f,0.f,0.f,0.f};
    v8f acc01 = acc00, acc10 = acc00, acc11 = acc00;

    auto stage = [&](int buf, int k0) {
        // A tile (128x32): two async b128 DMAs per thread
        unsigned l0 = (unsigned)(unsigned long long)&As[buf][smm * LPAD + sch];
        unsigned l1 = (unsigned)(unsigned long long)&As[buf][(smm + 64) * LPAD + sch];
        unsigned long long g0 =
            (unsigned long long)&A[(size_t)(m0 + smm) * lda + k0 + sch];
        unsigned long long g1 =
            (unsigned long long)&A[(size_t)(m0 + smm + 64) * lda + k0 + sch];
        ASYNC_B128(l0, g0);
        ASYNC_B128(l1, g1);
        if (TRANSB) {
            // B stored (N x K): contiguous in K -> async DMA straight in
            unsigned lb = (unsigned)(unsigned long long)&Bs[buf][smm * LPAD + sch];
            unsigned long long gb =
                (unsigned long long)&B[(size_t)(n0 + smm) * ldb + k0 + sch];
            ASYNC_B128(lb, gb);
        } else {
            // B stored (K x N): transpose through VGPRs into Bs[n][k]
            uint4 d = *(const uint4*)&B[(size_t)(k0 + bkk) * ldb + n0 + bnb];
            bf16_t tmp[8]; *(uint4*)tmp = d;
#pragma unroll
            for (int q = 0; q < 8; ++q) Bs[buf][(bnb + q) * LPAD + bkk] = tmp[q];
        }
    };

    const int nsteps = K / BK;
    stage(0, 0);
    WAIT_ASYNC();
    __syncthreads();

    for (int s = 0; s < nsteps; ++s) {
        const int buf = s & 1;
        if (s + 1 < nsteps) stage(buf ^ 1, (s + 1) * BK);

        // ---- fragments: contiguous b128 loads per ISA 7.12.2 ----
        // A frag lane data: K in [8*hi, 8*hi+7] and [16+8*hi, 16+8*hi+7]
        const int ar0 = (wm * 32 + r) * LPAD;
        const int ar1 = (wm * 32 + 16 + r) * LPAD;
        v8bf a0l = *(const v8bf*)&As[buf][ar0 + 8 * hi];
        v8bf a0h = *(const v8bf*)&As[buf][ar0 + 16 + 8 * hi];
        v8bf a1l = *(const v8bf*)&As[buf][ar1 + 8 * hi];
        v8bf a1h = *(const v8bf*)&As[buf][ar1 + 16 + 8 * hi];
        // B frag lane data: K in [8*hi, 8*hi+7] and [16+8*hi, 16+8*hi+7] at fixed column
        const int br0 = (wn * 32 + r) * LPAD + 8 * hi;
        const int br1 = (wn * 32 + 16 + r) * LPAD + 8 * hi;
        v8bf b0l = *(const v8bf*)&Bs[buf][br0];
        v8bf b0h = *(const v8bf*)&Bs[buf][br0 + 16];
        v8bf b1l = *(const v8bf*)&Bs[buf][br1];
        v8bf b1h = *(const v8bf*)&Bs[buf][br1 + 16];

        v16bf a0 = __builtin_shufflevector(a0l, a0h, 0,1,2,3,4,5,6,7,8,9,10,11,12,13,14,15);
        v16bf a1 = __builtin_shufflevector(a1l, a1h, 0,1,2,3,4,5,6,7,8,9,10,11,12,13,14,15);
        v16bf b0 = __builtin_shufflevector(b0l, b0h, 0,1,2,3,4,5,6,7,8,9,10,11,12,13,14,15);
        v16bf b1 = __builtin_shufflevector(b1l, b1h, 0,1,2,3,4,5,6,7,8,9,10,11,12,13,14,15);

        acc00 = WMMA_BF16(a0, b0, acc00);
        acc01 = WMMA_BF16(a0, b1, acc01);
        acc10 = WMMA_BF16(a1, b0, acc10);
        acc11 = WMMA_BF16(a1, b1, acc11);

        WAIT_ASYNC();          // next tile fully in LDS
        __syncthreads();       // and everyone done reading this tile
    }

    // C/D layout: VGPR i -> M = i + 8*hi, N = lane%16
#pragma unroll
    for (int i = 0; i < 8; ++i) {
        const int mA = m0 + wm * 32 + i + 8 * hi;
        const int mB = mA + 16;
        const int nA = n0 + wn * 32 + r;
        const int nB = nA + 16;
        size_t i00 = (size_t)mA * ldc + nA;
        size_t i01 = (size_t)mA * ldc + nB;
        size_t i10 = (size_t)mB * ldc + nA;
        size_t i11 = (size_t)mB * ldc + nB;
        if (accum) {
            C[i00] += alpha * acc00[i]; C[i01] += alpha * acc01[i];
            C[i10] += alpha * acc10[i]; C[i11] += alpha * acc11[i];
        } else {
            C[i00] = alpha * acc00[i]; C[i01] = alpha * acc01[i];
            C[i10] = alpha * acc10[i]; C[i11] = alpha * acc11[i];
        }
    }
}

// ============================================================
// Elementwise / normalization / attention support kernels
// ============================================================
__global__ __launch_bounds__(256) void cvt_bf16(const float* __restrict__ in,
                                                bf16_t* __restrict__ out, long n)
{
    for (long i = (long)blockIdx.x * 256 + threadIdx.x; i < n;
         i += (long)gridDim.x * 256)
        out[i] = to_bf(in[i]);
}

__global__ __launch_bounds__(256) void rmsnorm_bf16(
    const float* __restrict__ x, const float* __restrict__ w,
    bf16_t* __restrict__ out, int dim)
{
    int t = blockIdx.x, tid = threadIdx.x;
    const float* row = x + (size_t)t * dim;
    __shared__ float red[256];
    float ss = 0.f;
    for (int c = tid; c < dim; c += 256) { float v = row[c]; ss += v * v; }
    red[tid] = ss; __syncthreads();
    for (int s2 = 128; s2 > 0; s2 >>= 1) {
        if (tid < s2) red[tid] += red[tid + s2];
        __syncthreads();
    }
    float r = rsqrtf(red[0] / dim + EPS_F);
    for (int c = tid; c < dim; c += 256)
        out[(size_t)t * dim + c] = to_bf(row[c] * r * w[c]);
}

// split qkv, apply RoPE (+KEY_MULT) and replicate KV heads to NHQ, write bf16
__global__ __launch_bounds__(256) void rope_split(
    const float* __restrict__ qkv, const int* __restrict__ pos,
    bf16_t* __restrict__ Qh, bf16_t* __restrict__ Kh, bf16_t* __restrict__ Vh)
{
    int t = blockIdx.x, tid = threadIdx.x;
    const float* row = qkv + (size_t)t * QKV_N;
    float p = (float)pos[t];
    for (int i = tid; i < NHQ * 64; i += 256) {
        int hq = i >> 6, d = i & 63;
        float inv = __expf(-9.210340372f * (2.f * d / 128.f));
        float ang = p * inv, c = __cosf(ang), s = __sinf(ang);
        float x1 = row[hq * 128 + d], x2 = row[hq * 128 + d + 64];
        bf16_t* qo = Qh + ((size_t)hq * T_SEQ + t) * DH;
        qo[d]      = to_bf(x1 * c - x2 * s);
        qo[d + 64] = to_bf(x2 * c + x1 * s);
    }
    for (int i = tid; i < NHKV * 64; i += 256) {
        int kh = i >> 6, d = i & 63;
        float inv = __expf(-9.210340372f * (2.f * d / 128.f));
        float ang = p * inv, c = __cosf(ang), s = __sinf(ang);
        float x1 = row[2048 + kh * 128 + d] * 0.5f;
        float x2 = row[2048 + kh * 128 + d + 64] * 0.5f;
        float o1 = x1 * c - x2 * s, o2 = x2 * c + x1 * s;
        for (int rp = 0; rp < 2; ++rp) {
            bf16_t* ko = Kh + ((size_t)(kh * 2 + rp) * T_SEQ + t) * DH;
            ko[d] = to_bf(o1); ko[d + 64] = to_bf(o2);
        }
    }
    for (int i = tid; i < NHKV * 128; i += 256) {
        int kh = i >> 7, d = i & 127;
        float v = row[3072 + kh * 128 + d];
        for (int rp = 0; rp < 2; ++rp)
            Vh[((size_t)(kh * 2 + rp) * T_SEQ + t) * DH + d] = to_bf(v);
    }
}

__global__ __launch_bounds__(256) void softmax_causal(
    const float* __restrict__ S, bf16_t* __restrict__ P, int Tn)
{
    int q = blockIdx.x, tid = threadIdx.x;
    const float* row = S + (size_t)q * Tn;
    __shared__ float red[256];
    float m = -3.4e38f;
    for (int c = tid; c <= q; c += 256) m = fmaxf(m, row[c]);
    red[tid] = m; __syncthreads();
    for (int s2 = 128; s2 > 0; s2 >>= 1) {
        if (tid < s2) red[tid] = fmaxf(red[tid], red[tid + s2]);
        __syncthreads();
    }
    m = red[0]; __syncthreads();
    float sum = 0.f;
    for (int c = tid; c <= q; c += 256) sum += __expf(row[c] - m);
    red[tid] = sum; __syncthreads();
    for (int s2 = 128; s2 > 0; s2 >>= 1) {
        if (tid < s2) red[tid] += red[tid + s2];
        __syncthreads();
    }
    float inv = 1.f / red[0];
    for (int c = tid; c < Tn; c += 256)
        P[(size_t)q * Tn + c] = (c <= q) ? to_bf(__expf(row[c] - m) * inv)
                                         : to_bf(0.f);
}

// ============================================================
// Mamba branch support kernels
// ============================================================
__global__ __launch_bounds__(256) void mup_scale(float* __restrict__ zx, long n)
{
    for (long i = (long)blockIdx.x * 256 + threadIdx.x; i < n;
         i += (long)gridDim.x * 256) {
        int c = (int)(i % ZX_N);
        float m = (c < D_SSM)             ? 1.0f
                : (c < 2 * D_SSM)         ? 0.8f
                : (c < 2 * D_SSM + GS)    ? 0.9f
                : (c < 2 * D_SSM + 2*GS)  ? 0.95f
                :                           1.1f;
        zx[i] *= m;
    }
}

__global__ __launch_bounds__(256) void conv_silu(
    const float* __restrict__ zx, const float* __restrict__ w,
    const float* __restrict__ b, float* __restrict__ out)
{
    long n = (long)T_SEQ * CONV_CH;
    for (long i = (long)blockIdx.x * 256 + threadIdx.x; i < n;
         i += (long)gridDim.x * 256) {
        int c = (int)(i % CONV_CH);
        int t = (int)(i / CONV_CH);
        float acc = b[c];
#pragma unroll
        for (int j = 0; j < 4; ++j) {
            int tt = t - 3 + j;
            if (tt >= 0) acc += w[c * 4 + j] * zx[(size_t)tt * ZX_N + D_SSM + c];
        }
        out[i] = siluf(acc);
    }
}

__global__ __launch_bounds__(256) void dt_prep(
    const float* __restrict__ zx, const float* __restrict__ dt_bias,
    const float* __restrict__ A_log, float* __restrict__ dtb,
    float* __restrict__ Aneg)
{
    long n = (long)T_SEQ * SSM_NH;
    for (long i = (long)blockIdx.x * 256 + threadIdx.x; i < n;
         i += (long)gridDim.x * 256) {
        int h = (int)(i % SSM_NH);
        int t = (int)(i / SSM_NH);
        float v = zx[(size_t)t * ZX_N + (2 * D_SSM + 2 * GS) + h] + dt_bias[h];
        dtb[i] = (v > 20.f) ? v : log1pf(__expf(v));
        if (t == 0) Aneg[h] = -__expf(A_log[h]);
    }
}

// selective scan: one wave per (head h, channel p); 128 state dims = 4/lane
__global__ __launch_bounds__(32) void ssm_scan(
    const float* __restrict__ xconv, const float* __restrict__ dtb,
    const float* __restrict__ Aneg, const float* __restrict__ Dp,
    float* __restrict__ y)
{
    int h = blockIdx.x >> 6;
    int p = blockIdx.x & 63;
    int lane = threadIdx.x;
    float hs0 = 0.f, hs1 = 0.f, hs2 = 0.f, hs3 = 0.f;
    float a = Aneg[h], dprm = Dp[h];
    for (int t = 0; t < T_SEQ; ++t) {
        const float* row = xconv + (size_t)t * CONV_CH;
        float dt = dtb[t * SSM_NH + h];
        float decay = __expf(dt * a);
        float x = row[h * HEADD + p];
        float coef = dt * x;
        const float* Brow = row + D_SSM;
        const float* Crow = row + D_SSM + GS;
        float part;
        hs0 = hs0 * decay + coef * Brow[lane];      part  = hs0 * Crow[lane];
        hs1 = hs1 * decay + coef * Brow[lane + 32]; part += hs1 * Crow[lane + 32];
        hs2 = hs2 * decay + coef * Brow[lane + 64]; part += hs2 * Crow[lane + 64];
        hs3 = hs3 * decay + coef * Brow[lane + 96]; part += hs3 * Crow[lane + 96];
        for (int off = 16; off > 0; off >>= 1) part += __shfl_down(part, off, 32);
        if (lane == 0) y[(size_t)t * D_SSM + h * HEADD + p] = part + dprm * x;
    }
}

__global__ __launch_bounds__(256) void ssm_gate_norm(
    const float* __restrict__ y, const float* __restrict__ zx,
    const float* __restrict__ nw, bf16_t* __restrict__ out)
{
    int t = blockIdx.x, tid = threadIdx.x;
    __shared__ float g[D_SSM];
    __shared__ float red[256];
    float ss = 0.f;
    for (int c = tid; c < D_SSM; c += 256) {
        float z = zx[(size_t)t * ZX_N + c];
        float gv = y[(size_t)t * D_SSM + c] * siluf(z);
        g[c] = gv; ss += gv * gv;
    }
    red[tid] = ss; __syncthreads();
    for (int s2 = 128; s2 > 0; s2 >>= 1) {
        if (tid < s2) red[tid] += red[tid + s2];
        __syncthreads();
    }
    float r = rsqrtf(red[0] / D_SSM + EPS_F);
    for (int c = tid; c < D_SSM; c += 256)
        out[(size_t)t * D_SSM + c] = to_bf(g[c] * r * nw[c]);
}

__global__ __launch_bounds__(256) void add2(const float* __restrict__ a,
                                            const float* __restrict__ b,
                                            float* __restrict__ o, long n)
{
    for (long i = (long)blockIdx.x * 256 + threadIdx.x; i < n;
         i += (long)gridDim.x * 256)
        o[i] = a[i] + b[i];
}

__global__ __launch_bounds__(256) void gate_silu_mul(
    const float* __restrict__ gu, bf16_t* __restrict__ out)
{
    long n = (long)T_SEQ * INTER;
    for (long i = (long)blockIdx.x * 256 + threadIdx.x; i < n;
         i += (long)gridDim.x * 256) {
        int  c = (int)(i % INTER);
        long t = i / INTER;
        float g = gu[t * (2 * INTER) + c];          // GATE_MULT = 1.0
        float u = gu[t * (2 * INTER) + INTER + c];
        out[i] = to_bf(siluf(g) * u);
    }
}

// ============================================================
// Host orchestration
// ============================================================
static inline dim3 gemm_grid(int M, int N) { return dim3(N / BN, M / BM, 1); }

extern "C" void kernel_launch(void* const* d_in, const int* in_sizes, int n_in,
                              void* d_out, int out_size, void* d_ws, size_t ws_size,
                              hipStream_t stream)
{
    (void)in_sizes; (void)n_in; (void)out_size; (void)ws_size;
    const float* hidden      = (const float*)d_in[0];
    const int*   positions   = (const int*)  d_in[1];
    const float* w_qkv       = (const float*)d_in[2];
    const float* w_o         = (const float*)d_in[3];
    const float* in_proj     = (const float*)d_in[4];
    const float* conv_w      = (const float*)d_in[5];
    const float* conv_b      = (const float*)d_in[6];
    const float* A_log       = (const float*)d_in[7];
    const float* dt_bias     = (const float*)d_in[8];
    const float* D_param     = (const float*)d_in[9];
    const float* ssm_norm_w  = (const float*)d_in[10];
    const float* out_proj    = (const float*)d_in[11];
    const float* input_ln_w  = (const float*)d_in[12];
    const float* pre_ff_ln_w = (const float*)d_in[13];
    const float* gate_up     = (const float*)d_in[14];
    const float* down        = (const float*)d_in[15];
    float* out = (float*)d_out;

    char* ws = (char*)d_ws;
    size_t off = 0;
    auto alloc = [&](size_t bytes) -> void* {
        void* p = ws + off;
        off += (bytes + 255) & ~(size_t)255;
        return p;
    };

    bf16_t* wqkv_b  = (bf16_t*)alloc((size_t)D_MODEL * QKV_N * 2);
    bf16_t* wo_b    = (bf16_t*)alloc((size_t)D_MODEL * D_MODEL * 2);
    bf16_t* winp_b  = (bf16_t*)alloc((size_t)D_MODEL * ZX_N * 2);
    bf16_t* woutp_b = (bf16_t*)alloc((size_t)D_SSM * D_MODEL * 2);
    bf16_t* wgu_b   = (bf16_t*)alloc((size_t)D_MODEL * 2 * INTER * 2);
    bf16_t* wdn_b   = (bf16_t*)alloc((size_t)INTER * D_MODEL * 2);
    bf16_t* h_b     = (bf16_t*)alloc((size_t)T_SEQ * D_MODEL * 2);
    float*  qkv_f   = (float*) alloc((size_t)T_SEQ * QKV_N * 4);
    bf16_t* Qh      = (bf16_t*)alloc((size_t)NHQ * T_SEQ * DH * 2);
    bf16_t* Kh      = (bf16_t*)alloc((size_t)NHQ * T_SEQ * DH * 2);
    bf16_t* Vh      = (bf16_t*)alloc((size_t)NHQ * T_SEQ * DH * 2);
    float*  S_f     = (float*) alloc((size_t)T_SEQ * T_SEQ * 4);
    bf16_t* P_b     = (bf16_t*)alloc((size_t)T_SEQ * T_SEQ * 2);
    float*  attn_f  = (float*) alloc((size_t)T_SEQ * D_MODEL * 4);
    bf16_t* attn_b  = (bf16_t*)alloc((size_t)T_SEQ * D_MODEL * 2);
    float*  zx_f    = (float*) alloc((size_t)T_SEQ * ZX_N * 4);
    float*  xconv_f = (float*) alloc((size_t)T_SEQ * CONV_CH * 4);
    float*  dtb_f   = (float*) alloc((size_t)T_SEQ * SSM_NH * 4);
    float*  Aneg_f  = (float*) alloc((size_t)SSM_NH * 4);
    float*  y_f     = (float*) alloc((size_t)T_SEQ * D_SSM * 4);
    bf16_t* yn_b    = (bf16_t*)alloc((size_t)T_SEQ * D_SSM * 2);
    float*  hsum_f  = (float*) alloc((size_t)T_SEQ * D_MODEL * 4);
    float*  resid_f = (float*) alloc((size_t)T_SEQ * D_MODEL * 4);
    bf16_t* h2_b    = (bf16_t*)alloc((size_t)T_SEQ * D_MODEL * 2);
    float*  gu_f    = (float*) alloc((size_t)T_SEQ * 2 * INTER * 4);
    bf16_t* mlp_b   = (bf16_t*)alloc((size_t)T_SEQ * INTER * 2);

    const int TPB = 256;

    // ---- weight fp32 -> bf16 (one pass over HBM) ----
    cvt_bf16<<<2048, TPB, 0, stream>>>(w_qkv,    wqkv_b,  (long)D_MODEL * QKV_N);
    cvt_bf16<<<2048, TPB, 0, stream>>>(w_o,      wo_b,    (long)D_MODEL * D_MODEL);
    cvt_bf16<<<2048, TPB, 0, stream>>>(in_proj,  winp_b,  (long)D_MODEL * ZX_N);
    cvt_bf16<<<2048, TPB, 0, stream>>>(out_proj, woutp_b, (long)D_SSM * D_MODEL);
    cvt_bf16<<<2048, TPB, 0, stream>>>(gate_up,  wgu_b,   (long)D_MODEL * 2 * INTER);
    cvt_bf16<<<2048, TPB, 0, stream>>>(down,     wdn_b,   (long)INTER * D_MODEL);

    // ---- shared input RMSNorm (ATTN_IN_MULT = SSM_IN_MULT = 1.0) ----
    rmsnorm_bf16<<<T_SEQ, TPB, 0, stream>>>(hidden, input_ln_w, h_b, D_MODEL);

    // ---- attention branch ----
    gemm_wmma_bf16<false><<<gemm_grid(T_SEQ, QKV_N), TPB, 0, stream>>>(
        h_b, D_MODEL, wqkv_b, QKV_N, qkv_f, QKV_N,
        T_SEQ, QKV_N, D_MODEL, 1.0f, 0);
    rope_split<<<T_SEQ, TPB, 0, stream>>>(qkv_f, positions, Qh, Kh, Vh);

    const float scale = 0.08838834764831845f; // 1/sqrt(128)
    for (int hq = 0; hq < NHQ; ++hq) {
        const bf16_t* Qp = Qh + (size_t)hq * T_SEQ * DH;
        const bf16_t* Kp = Kh + (size_t)hq * T_SEQ * DH;
        const bf16_t* Vp = Vh + (size_t)hq * T_SEQ * DH;
        gemm_wmma_bf16<true><<<gemm_grid(T_SEQ, T_SEQ), TPB, 0, stream>>>(
            Qp, DH, Kp, DH, S_f, T_SEQ, T_SEQ, T_SEQ, DH, scale, 0);
        softmax_causal<<<T_SEQ, TPB, 0, stream>>>(S_f, P_b, T_SEQ);
        gemm_wmma_bf16<false><<<gemm_grid(T_SEQ, DH), TPB, 0, stream>>>(
            P_b, T_SEQ, Vp, DH, attn_f + hq * DH, D_MODEL,
            T_SEQ, DH, T_SEQ, 1.0f, 0);
    }
    cvt_bf16<<<2048, TPB, 0, stream>>>(attn_f, attn_b, (long)T_SEQ * D_MODEL);
    gemm_wmma_bf16<false><<<gemm_grid(T_SEQ, D_MODEL), TPB, 0, stream>>>(
        attn_b, D_MODEL, wo_b, D_MODEL, hsum_f, D_MODEL,
        T_SEQ, D_MODEL, D_MODEL, 0.5f, 0);            // ATTN_OUT_MULT

    // ---- Mamba branch ----
    gemm_wmma_bf16<false><<<gemm_grid(T_SEQ, ZX_N), TPB, 0, stream>>>(
        h_b, D_MODEL, winp_b, ZX_N, zx_f, ZX_N,
        T_SEQ, ZX_N, D_MODEL, 1.0f, 0);
    mup_scale<<<2048, TPB, 0, stream>>>(zx_f, (long)T_SEQ * ZX_N);
    conv_silu<<<2048, TPB, 0, stream>>>(zx_f, conv_w, conv_b, xconv_f);
    dt_prep<<<512, TPB, 0, stream>>>(zx_f, dt_bias, A_log, dtb_f, Aneg_f);
    ssm_scan<<<SSM_NH * HEADD, 32, 0, stream>>>(xconv_f, dtb_f, Aneg_f,
                                                D_param, y_f);
    ssm_gate_norm<<<T_SEQ, TPB, 0, stream>>>(y_f, zx_f, ssm_norm_w, yn_b);
    gemm_wmma_bf16<false><<<gemm_grid(T_SEQ, D_MODEL), TPB, 0, stream>>>(
        yn_b, D_SSM, woutp_b, D_MODEL, hsum_f, D_MODEL,
        T_SEQ, D_MODEL, D_SSM, 0.5f, 1);              // SSM_OUT_MULT, accumulate

    // ---- residual + MLP ----
    add2<<<2048, TPB, 0, stream>>>(hidden, hsum_f, resid_f,
                                   (long)T_SEQ * D_MODEL);
    rmsnorm_bf16<<<T_SEQ, TPB, 0, stream>>>(resid_f, pre_ff_ln_w, h2_b, D_MODEL);
    gemm_wmma_bf16<false><<<gemm_grid(T_SEQ, 2 * INTER), TPB, 0, stream>>>(
        h2_b, D_MODEL, wgu_b, 2 * INTER, gu_f, 2 * INTER,
        T_SEQ, 2 * INTER, D_MODEL, 1.0f, 0);
    gate_silu_mul<<<4096, TPB, 0, stream>>>(gu_f, mlp_b);
    hipMemcpyAsync(out, resid_f, (size_t)T_SEQ * D_MODEL * 4,
                   hipMemcpyDeviceToDevice, stream);
    gemm_wmma_bf16<false><<<gemm_grid(T_SEQ, D_MODEL), TPB, 0, stream>>>(
        mlp_b, INTER, wdn_b, D_MODEL, out, D_MODEL,
        T_SEQ, D_MODEL, INTER, 0.5f, 1);              // DOWN_MULT, accumulate
}
